// BayesianLinear_20151986553504
// MI455X (gfx1250) — matrix operand order repaired and checked by reading, not compiled, not run on hardware
//
#include <hip/hip_runtime.h>

#define IN_F  4096
#define OUT_F 4096
#define BATCH 8192

typedef __bf16 v16bf __attribute__((ext_vector_type(16)));
typedef __bf16 v4bf  __attribute__((ext_vector_type(4)));
typedef float  v8f   __attribute__((ext_vector_type(8)));
typedef int    v4i   __attribute__((vector_size(16)));

#if __has_builtin(__builtin_amdgcn_global_load_async_to_lds_b128)
#define HAVE_ASYNC_LDS 1
// builtin signature (from hipcc diagnostic): param0 = v4i __device__* (AS1)
#define AS1V(p) ((__attribute__((address_space(1))) v4i*)(p))
#define AS3V(p) ((__attribute__((address_space(3))) v4i*)(p))
#endif

__device__ __forceinline__ void wait_async_zero() {
#if defined(HAVE_ASYNC_LDS)
#if __has_builtin(__builtin_amdgcn_s_wait_asynccnt)
  __builtin_amdgcn_s_wait_asynccnt(0);
#else
  asm volatile("s_wait_asynccnt 0" ::: "memory");
#endif
#endif
}

// ---------------------------------------------------------------------------
// Prep 1: sample weight = mu + eps * exp(0.5*log_var), pack f32 -> bf16 AND
// transpose: Wt[k][n] = sample(W[n][k]).  One-time pass, LDS-tiled so both
// global reads (along k) and writes (along n) are coalesced.
// ---------------------------------------------------------------------------
__global__ void __launch_bounds__(256) prep_weight_t_bf16(
    const float* __restrict__ mu, const float* __restrict__ lv,
    const float* __restrict__ eps, __bf16* __restrict__ Wt)
{
  __shared__ __align__(16) __bf16 T[64][76];   // [n_local][k_local], padded
  const int n0 = blockIdx.x * 64;              // source row block (out feature)
  const int k0 = blockIdx.y * 64;              // source col block (in feature)
  const int tid = threadIdx.x;
  const int kl4 = (tid & 15) * 4;              // k_local, step 4
  const int nl  = tid >> 4;                    // 0..15

  #pragma unroll
  for (int j = 0; j < 4; ++j) {
    int row = nl + j * 16;                     // n_local 0..63
    size_t idx = (size_t)(n0 + row) * IN_F + k0 + kl4;
    float4 m = *(const float4*)(mu  + idx);
    float4 l = *(const float4*)(lv  + idx);
    float4 e = *(const float4*)(eps + idx);
    v4bf h;
    h.x = (__bf16)fmaf(e.x, __expf(0.5f * l.x), m.x);
    h.y = (__bf16)fmaf(e.y, __expf(0.5f * l.y), m.y);
    h.z = (__bf16)fmaf(e.z, __expf(0.5f * l.z), m.z);
    h.w = (__bf16)fmaf(e.w, __expf(0.5f * l.w), m.w);
    *(v4bf*)(&T[row][kl4]) = h;
  }
  __syncthreads();

  #pragma unroll
  for (int j = 0; j < 2; ++j) {
    int chunk = tid + 256 * j;                 // 512 x 16B chunks
    int kl = chunk >> 3;                       // 0..63
    int nc = (chunk & 7) * 8;                  // 0..56
    union { uint4 u; __bf16 h[8]; } o;
    #pragma unroll
    for (int t = 0; t < 8; ++t) o.h[t] = T[nc + t][kl];
    *(uint4*)(Wt + (size_t)(k0 + kl) * OUT_F + n0 + nc) = o.u;  // coalesced
  }
}

// Prep 2: sampled bias in f32 (kept full precision for the epilogue).
__global__ void __launch_bounds__(256) prep_bias_f32(
    const float* __restrict__ mu, const float* __restrict__ lv,
    const float* __restrict__ eps, float* __restrict__ out)
{
  int i = blockIdx.x * 256 + threadIdx.x;
  out[i] = fmaf(eps[i], __expf(0.5f * lv[i]), mu[i]);
}

// ---------------------------------------------------------------------------
// Main GEMM: out[b,o] = sum_k Xbf16[b,k] * Wt[k,o] + bias[o]
// Block = 256 threads = 8 waves (2 x 4), WG tile 128(M) x 256(N),
// wave tile 64x64 = 16 v_wmma_f32_16x16x32_bf16 per K-step.
// Double-buffered LDS pipeline, one barrier per K-step:
//   stage(i+1) issues async W DMA (ASYNCcnt) + X f32->bf16 staging, then
//   overlaps with compute(i); s_wait_asynccnt 0 only waits for the buffer
//   that is about to be consumed.
// ---------------------------------------------------------------------------
__global__ void __launch_bounds__(256) bayes_gemm_wmma(
    const float* __restrict__ X, const __bf16* __restrict__ Wt,
    const float* __restrict__ bias, float* __restrict__ out)
{
  constexpr int K = IN_F, N = OUT_F, KB = 32, STEPS = K / KB;
  __shared__ __align__(16) __bf16 Xs[2][128][40];  // pad 32->40: 80B stride
  __shared__ __align__(16) __bf16 Ws[2][32][264];  // [k][n], pad 256->264

  const int tid  = threadIdx.x;
  const int lane = tid & 31;
  const int wave = tid >> 5;
  const int wm = wave >> 2;          // 0..1  (M direction)
  const int wn = wave & 3;           // 0..3  (N direction)
  const int lr = lane & 15;
  const int lh = lane >> 4;
  const int mbase = blockIdx.y * 128;
  const int nbase = blockIdx.x * 256;

  auto stage = [&](int kpos, int buf) {
    // ---- W tile: contiguous copy Wt[kpos..+31][nbase..+255] ----
#if defined(HAVE_ASYNC_LDS)
    #pragma unroll
    for (int j = 0; j < 4; ++j) {
      int chunk = tid + 256 * j;               // 1024 x 16B chunks
      int kl = chunk >> 5;                     // 0..31
      int nc = (chunk & 31) * 8;               // 0..248
      const __bf16* g = Wt + (size_t)(kpos + kl) * N + nbase + nc;
      __builtin_amdgcn_global_load_async_to_lds_b128(
          AS1V(g), AS3V(&Ws[buf][kl][nc]), 0, 0);
    }
#else
    #pragma unroll
    for (int j = 0; j < 4; ++j) {
      int chunk = tid + 256 * j;
      int kl = chunk >> 5;
      int nc = (chunk & 31) * 8;
      uint4 v = *(const uint4*)(Wt + (size_t)(kpos + kl) * N + nbase + nc);
      *(uint4*)(&Ws[buf][kl][nc]) = v;
    }
#endif
    // ---- X tile: 128 rows x 32 K, f32 -> bf16 ----
    #pragma unroll
    for (int j = 0; j < 4; ++j) {
      int chunk = tid + 256 * j;               // 1024 float4 chunks
      int row   = chunk >> 3;
      int kc    = (chunk & 7) * 4;
      float4 v  = *(const float4*)(X + (size_t)(mbase + row) * K + kpos + kc);
      v4bf h;
      h.x = (__bf16)v.x; h.y = (__bf16)v.y; h.z = (__bf16)v.z; h.w = (__bf16)v.w;
      *(v4bf*)(&Xs[buf][row][kc]) = h;
    }
  };

  v8f acc[4][4] = {};
  stage(0, 0);

  for (int i = 0; i < STEPS; ++i) {
    const int buf = i & 1;
    wait_async_zero();     // this wave's stage(i) DMA has landed in LDS
    __syncthreads();       // all waves staged buf; all done reading buf^1

    if (i + 1 < STEPS) stage((i + 1) * KB, buf ^ 1);   // overlaps compute(i)

    // ---- fragment loads (ISA 7.12.2 layouts) ----
    union Frag { uint4 u[2]; v16bf v; } a[4], b[4];
    #pragma unroll
    for (int r = 0; r < 4; ++r) {
      // A: lane holds row lr, K chunks {8*lh..} and {16+8*lh..}
      const __bf16* p = &Xs[buf][wm * 64 + r * 16 + lr][8 * lh];
      a[r].u[0] = *(const uint4*)p;
      a[r].u[1] = *(const uint4*)(p + 16);
    }
    #pragma unroll
    for (int c = 0; c < 4; ++c) {
      // B: lane holds column lr, K chunks {8*lh..} and {16+8*lh..}
      const __bf16* p = &Ws[buf][8 * lh][wn * 64 + c * 16 + lr];
      #pragma unroll
      for (int s = 0; s < 16; ++s)
        b[c].v[s] = p[(s < 8 ? s : 8 + s) * 264];
    }

    #pragma unroll
    for (int r = 0; r < 4; ++r)
      #pragma unroll
      for (int c = 0; c < 4; ++c)
        acc[r][c] = __builtin_amdgcn_wmma_f32_16x16x32_bf16(
            false, a[r].v, false, b[c].v, (short)0, acc[r][c], false, false);
  }

  // ---- epilogue: C frag row = vgpr + 8*lh, col = lr ----
  #pragma unroll
  for (int c = 0; c < 4; ++c) {
    const int col = nbase + wn * 64 + c * 16 + lr;
    const float bv = bias[col];
    #pragma unroll
    for (int r = 0; r < 4; ++r) {
      const int rw = mbase + wm * 64 + r * 16 + 8 * lh;
      #pragma unroll
      for (int i = 0; i < 8; ++i)
        out[(size_t)(rw + i) * N + col] = acc[r][c][i] + bv;
    }
  }
}

// ---------------------------------------------------------------------------
// Fallback (only if workspace is too small for the bf16 weight pack).
// ---------------------------------------------------------------------------
__global__ void __launch_bounds__(256) bayes_naive(
    const float* __restrict__ x,  const float* __restrict__ wmu,
    const float* __restrict__ wlv, const float* __restrict__ bmu,
    const float* __restrict__ blv, const float* __restrict__ weps,
    const float* __restrict__ beps, float* __restrict__ out)
{
  int o = blockIdx.x * 256 + threadIdx.x;
  int b = blockIdx.y;
  float acc = 0.f;
  for (int i = 0; i < IN_F; ++i) {
    float w = fmaf(weps[(size_t)o * IN_F + i],
                   __expf(0.5f * wlv[(size_t)o * IN_F + i]),
                   wmu[(size_t)o * IN_F + i]);
    acc = fmaf(x[(size_t)b * IN_F + i], w, acc);
  }
  out[(size_t)b * OUT_F + o] = acc + fmaf(beps[o], __expf(0.5f * blv[o]), bmu[o]);
}

// ---------------------------------------------------------------------------
extern "C" void kernel_launch(void* const* d_in, const int* in_sizes, int n_in,
                              void* d_out, int out_size, void* d_ws, size_t ws_size,
                              hipStream_t stream) {
  (void)in_sizes; (void)n_in; (void)out_size;
  const float* x    = (const float*)d_in[0];
  const float* wmu  = (const float*)d_in[1];
  const float* wlv  = (const float*)d_in[2];
  const float* bmu  = (const float*)d_in[3];
  const float* blv  = (const float*)d_in[4];
  const float* weps = (const float*)d_in[5];
  const float* beps = (const float*)d_in[6];
  float* out = (float*)d_out;

  const size_t w_bytes    = (size_t)OUT_F * IN_F * sizeof(__bf16); // 32 MiB
  const size_t bias_bytes = (size_t)OUT_F * sizeof(float);
  if (ws_size < w_bytes + bias_bytes) {
    bayes_naive<<<dim3(OUT_F / 256, BATCH), 256, 0, stream>>>(
        x, wmu, wlv, bmu, blv, weps, beps, out);
    return;
  }

  __bf16* Wt   = (__bf16*)d_ws;            // transposed sampled weight [k][n]
  float* biasf = (float*)((char*)d_ws + w_bytes);

  prep_weight_t_bf16<<<dim3(OUT_F / 64, IN_F / 64), 256, 0, stream>>>(
      wmu, wlv, weps, Wt);
  prep_bias_f32<<<OUT_F / 256, 256, 0, stream>>>(bmu, blv, beps, biasf);

  bayes_gemm_wmma<<<dim3(OUT_F / 256, BATCH / 128), 256, 0, stream>>>(
      x, Wt, biasf, out);
}
